// GCNCommunityModel_33552284516503
// MI455X (gfx1250) — matrix operand order repaired and checked by reading, not compiled, not run on hardware
//
#include <hip/hip_runtime.h>
#include <hip/hip_bf16.h>

typedef __attribute__((ext_vector_type(16))) __bf16 v16bf;
typedef __attribute__((ext_vector_type(8)))  float  v8f;

#define CDIV(a, b) (((a) + (b) - 1) / (b))

// ---------------------------------------------------------------- utilities

__global__ void k_zero(float* __restrict__ p, size_t n) {
  size_t i = (size_t)blockIdx.x * blockDim.x + threadIdx.x;
  if (i < n) p[i] = 0.0f;
}

__global__ void k_relu(float* __restrict__ x, size_t n) {
  size_t i = (size_t)blockIdx.x * blockDim.x + threadIdx.x;
  if (i < n) x[i] = fmaxf(x[i], 0.0f);
}

// Pack fp32 weight W[Kd,F] into bf16 in the exact CDNA5 WMMA B-operand layout:
// Wp[((kb*ntiles + ntile)*32 + lane)*16 + j] = W[(kb*32 + (j<8 ? (lane>>4)*8 + j : 16 + (lane>>4)*8 + (j-8)))*F + ntile*16 + (lane&15)]
// so each wave fetches its whole B fragment as one aligned 32B vector load.
__global__ void k_pack_w(const float* __restrict__ W, __bf16* __restrict__ Wp, int Kd, int F) {
  int i = blockIdx.x * blockDim.x + threadIdx.x;
  int total = Kd * F;
  if (i >= total) return;
  int j = i & 15;
  int lane = (i >> 4) & 31;
  int t = i >> 9;
  int ntiles = F >> 4;
  int ntile = t % ntiles;
  int kbIdx = t / ntiles;
  int col = ntile * 16 + (lane & 15);
  int k = kbIdx * 32 + ((j < 8) ? ((lane >> 4) * 8 + j) : (16 + (lane >> 4) * 8 + (j - 8)));
  Wp[i] = (__bf16)W[(size_t)k * F + col];
}

// ---------------------------------------------------------------- degrees / norms

__global__ void k_deg(const int* __restrict__ dst, float* __restrict__ deg_in, int E) {
  int i = blockIdx.x * blockDim.x + threadIdx.x;
  if (i < E) atomicAdd(&deg_in[dst[i]], 1.0f);
}

__global__ void k_norm(const float* __restrict__ deg_in, float* __restrict__ deg,
                       float* __restrict__ snorm, float* __restrict__ pvec,
                       int N, float invE) {
  int i = blockIdx.x * blockDim.x + threadIdx.x;
  if (i < N) {
    float d = deg_in[i] + 1.0f;
    deg[i]   = d;
    snorm[i] = 1.0f / d;
    pvec[i]  = deg_in[i] * invE;
  }
}

__global__ void k_enorm(const int* __restrict__ src, const int* __restrict__ dst,
                        const float* __restrict__ deg, float* __restrict__ enorm, int E) {
  int i = blockIdx.x * blockDim.x + threadIdx.x;
  if (i < E) enorm[i] = rsqrtf(deg[src[i]] * deg[dst[i]]);
}

// ---------------------------------------------------------------- WMMA GEMMs
// A layout (16x32 bf16): lane holds row = lane&15; halves 0-7 -> K = hi*8+j,
//                        halves 8-15 -> K = 16+hi*8+(j-8).
// B layout (32x16 bf16): lane holds col = lane&15; half j -> K as in A (pre-packed).
// C layout: VGPR r -> M = m0 + hi*8 + r, N = col.
// A tile is staged in LDS in per-lane layout (bf16) by all waves cooperatively,
// so each wave's fragment is a single aligned 32B ds read, and global A traffic
// is not duplicated across the block's 8 N-tiles.

// Generic Y[M,F] = X[M,Kd] @ Wp. grid.x = M/16, block = (32, F/16).
__global__ void k_gemm(const float* __restrict__ X, const __bf16* __restrict__ Wp,
                       float* __restrict__ Y, int Kd, int F) {
  __shared__ alignas(32) __bf16 As[16 * 256];
  const int lane = threadIdx.x;
  const int wv = threadIdx.y;
  const int m0 = blockIdx.x * 16;
  const int tid = wv * 32 + lane;
  const int T = blockDim.y * 32;
  const int ntiles = F >> 4;

  for (int p = tid; p < (Kd << 4); p += T) {
    int j = p & 15;
    int ln = (p >> 4) & 31;
    int kb = p >> 9;
    int hi = ln >> 4;
    int jj = (j < 8) ? (hi * 8 + j) : (16 + hi * 8 + (j - 8));
    As[p] = (__bf16)X[(size_t)(m0 + (ln & 15)) * Kd + kb * 32 + jj];
  }
  __syncthreads();

  v8f acc = {};
  const int nkb = Kd >> 5;
  for (int kb = 0; kb < nkb; ++kb) {
    v16bf a = *(const v16bf*)(As + (((kb << 5) + lane) << 4));
    const __bf16* wp = Wp + ((size_t)((kb * ntiles + wv) << 5) + lane) * 16;
    if (kb + 1 < nkb) __builtin_prefetch(wp + (size_t)ntiles * 512, 0, 1);
    v16bf b = *(const v16bf*)wp;
    acc = __builtin_amdgcn_wmma_f32_16x16x32_bf16(false, a, false, b, (short)0, acc,
                                                  false, false);
  }

  const int hi = lane >> 4;
  float* yr = Y + (size_t)(m0 + hi * 8) * F + wv * 16 + (lane & 15);
#pragma unroll
  for (int r = 0; r < 8; ++r) yr[(size_t)r * F] = acc[r];
}

// Embed: rows = B*N, A = concat(rfp[row,:128], nf[row%N,:128]), Kd = 256,
// epilogue: * gamma/sqrt(1+eps) + beta.  grid.x = B*N/16, block = (32, 8).
__global__ void k_embed_gemm(const float* __restrict__ rfp, const float* __restrict__ nf,
                             const __bf16* __restrict__ Wp, const float* __restrict__ gamma,
                             const float* __restrict__ beta, float* __restrict__ Y, int N) {
  __shared__ alignas(32) __bf16 As[16 * 256];
  const int lane = threadIdx.x;
  const int wv = threadIdx.y;
  const int m0 = blockIdx.x * 16;
  const int tid = wv * 32 + lane;

  for (int p = tid; p < (256 << 4); p += 256) {
    int j = p & 15;
    int ln = (p >> 4) & 31;
    int kb = p >> 9;
    int hi = ln >> 4;
    int jj = (j < 8) ? (hi * 8 + j) : (16 + hi * 8 + (j - 8));
    int k = kb * 32 + jj;
    int row = m0 + (ln & 15);
    float v = (k < 128) ? rfp[(size_t)row * 128 + k]
                        : nf[(size_t)(row % N) * 128 + (k - 128)];
    As[p] = (__bf16)v;
  }
  __syncthreads();

  v8f acc = {};
  for (int kb = 0; kb < 8; ++kb) {
    v16bf a = *(const v16bf*)(As + (((kb << 5) + lane) << 4));
    v16bf b = *(const v16bf*)(Wp + ((size_t)((kb * 8 + wv) << 5) + lane) * 16);
    acc = __builtin_amdgcn_wmma_f32_16x16x32_bf16(false, a, false, b, (short)0, acc,
                                                  false, false);
  }

  const int hi = lane >> 4;
  const int col = wv * 16 + (lane & 15);
  const float sc = gamma[col] * rsqrtf(1.0f + 1e-3f);
  const float bt = beta[col];
  float* yr = Y + (size_t)(m0 + hi * 8) * 128 + col;
#pragma unroll
  for (int r = 0; r < 8; ++r) yr[(size_t)r * 128] = acc[r] * sc + bt;
}

// Pooled features: out[128,64] = S^T[128,N] @ enc[N,64]. grid = (8,4), block = 32.
__global__ void k_pool_gemm(const float* __restrict__ S, const float* __restrict__ enc,
                            float* __restrict__ out, int N) {
  const int lane = threadIdx.x;
  const int m0 = blockIdx.x * 16;
  const int n0 = blockIdx.y * 16;
  const int l15 = lane & 15;
  const int hi = lane >> 4;
  const int row = m0 + l15;
  const int col = n0 + l15;

  v8f acc = {};
  for (int kb = 0; kb < N; kb += 32) {
    v16bf a, b;
    const float* sp = S + (size_t)(kb + hi * 8) * 128 + row;  // A[m,kk] = S[kk,m]
#pragma unroll
    for (int j = 0; j < 8; ++j) {
      a[j]     = (__bf16)sp[(size_t)j * 128];
      a[8 + j] = (__bf16)sp[(size_t)(16 + j) * 128];
    }
    const float* bp = enc + (size_t)(kb + hi * 8) * 64 + col;
#pragma unroll
    for (int j = 0; j < 16; ++j) b[j] = (__bf16)bp[(size_t)(j + (j & 8)) * 64];
    acc = __builtin_amdgcn_wmma_f32_16x16x32_bf16(false, a, false, b, (short)0, acc,
                                                  false, false);
  }
  float* yr = out + (size_t)(m0 + hi * 8) * 64 + col;
#pragma unroll
  for (int r = 0; r < 8; ++r) yr[(size_t)r * 64] = acc[r];
}

// ---------------------------------------------------------------- GCN edge phase

__global__ void k_self_init4(const float* __restrict__ h, const float* __restrict__ snorm,
                             float* __restrict__ y, size_t tot4, int F4) {
  size_t i = (size_t)blockIdx.x * blockDim.x + threadIdx.x;
  if (i < tot4) {
    float s = snorm[i / (size_t)F4];
    float4 v = ((const float4*)h)[i];
    v.x *= s; v.y *= s; v.z *= s; v.w *= s;
    ((float4*)y)[i] = v;
  }
}

__global__ void k_scatter4(const int* __restrict__ src, const int* __restrict__ dst,
                           const float* __restrict__ enorm, const float* __restrict__ h,
                           float* __restrict__ y, int E, int F4) {
  size_t total = (size_t)E * F4;
  size_t stride = (size_t)gridDim.x * blockDim.x;
  int F = F4 * 4;
  for (size_t i = (size_t)blockIdx.x * blockDim.x + threadIdx.x; i < total; i += stride) {
    size_t e = i / (size_t)F4;
    int f = (int)(i - e * (size_t)F4) * 4;
    float en = enorm[e];
    float4 hv = *(const float4*)(h + (size_t)src[e] * F + f);
    float* yp = y + (size_t)dst[e] * F + f;
    atomicAdd(yp + 0, hv.x * en);
    atomicAdd(yp + 1, hv.y * en);
    atomicAdd(yp + 2, hv.z * en);
    atomicAdd(yp + 3, hv.w * en);
  }
}

// ---------------------------------------------------------------- elementwise glue

__global__ void k_sum_traj(const float* __restrict__ t, float* __restrict__ s,
                           size_t NF, int B) {
  size_t i = (size_t)blockIdx.x * blockDim.x + threadIdx.x;
  if (i < NF) {
    float acc = 0.0f;
    for (int b = 0; b < B; ++b) acc += t[(size_t)b * NF + i];
    s[i] = acc;
  }
}

__global__ void k_add_bcast(float* __restrict__ t, const float* __restrict__ s,
                            size_t NF, size_t bstride, int B, int dorelu) {
  size_t i = (size_t)blockIdx.x * blockDim.x + threadIdx.x;
  if (i < NF) {
    float sv = s[i];  // s already relu'ed when required
    for (int b = 0; b < B; ++b) {
      float v = t[(size_t)b * bstride + i];
      if (dorelu) v = fmaxf(v, 0.0f);
      t[(size_t)b * bstride + i] = v + sv;
    }
  }
}

__global__ void k_finalize(const float* __restrict__ t, const float* __restrict__ s,
                           float* __restrict__ enc, size_t NF, size_t bstride, int B) {
  size_t i = (size_t)blockIdx.x * blockDim.x + threadIdx.x;
  if (i < NF) {
    float acc = (float)B * s[i];
    for (int b = 0; b < B; ++b) acc += t[(size_t)b * bstride + i];
    enc[i] = acc;
  }
}

// ---------------------------------------------------------------- softmax / loss

__global__ void k_softmax128(float* __restrict__ S) {
  __shared__ float red[128];
  int t = threadIdx.x;
  size_t base = (size_t)blockIdx.x * 128;
  float v = S[base + t];
  red[t] = v;
  __syncthreads();
  for (int s = 64; s > 0; s >>= 1) {
    if (t < s) red[t] = fmaxf(red[t], red[t + s]);
    __syncthreads();
  }
  float m = red[0];
  __syncthreads();
  float e = expf(v - m);
  red[t] = e;
  __syncthreads();
  for (int s = 64; s > 0; s >>= 1) {
    if (t < s) red[t] += red[t + s];
    __syncthreads();
  }
  S[base + t] = e / red[0];
}

__global__ void k_diag(const int* __restrict__ src, const int* __restrict__ dst,
                       const float* __restrict__ S, float* __restrict__ diag, int E) {
  int k = threadIdx.x;  // 128
  float acc = 0.0f;
  for (int e = blockIdx.x; e < E; e += gridDim.x)
    acc += S[(size_t)src[e] * 128 + k] * S[(size_t)dst[e] * 128 + k];
  atomicAdd(&diag[k], acc);
}

__global__ void k_pk(const float* __restrict__ S, const float* __restrict__ p,
                     float* __restrict__ pk, int N) {
  int k = threadIdx.x;  // 128
  float acc = 0.0f;
  for (int n = blockIdx.x; n < N; n += gridDim.x)
    acc += S[(size_t)n * 128 + k] * p[n];
  atomicAdd(&pk[k], acc);
}

__global__ void k_plogp_sum(const float* __restrict__ p, float* __restrict__ scal, int N) {
  __shared__ float red[256];
  float acc = 0.0f;
  for (int i = blockIdx.x * blockDim.x + threadIdx.x; i < N; i += gridDim.x * blockDim.x)
    acc += p[i] * logf(p[i] + 1e-12f);
  red[threadIdx.x] = acc;
  __syncthreads();
  for (int s = 128; s > 0; s >>= 1) {
    if ((int)threadIdx.x < s) red[threadIdx.x] += red[threadIdx.x + s];
    __syncthreads();
  }
  if (threadIdx.x == 0) atomicAdd(&scal[0], red[0]);
}

__global__ void k_loss(const float* __restrict__ pk, const float* __restrict__ diag,
                       const float* __restrict__ scal, float* __restrict__ out, float invE) {
  __shared__ float r0[128], r1[128], r2[128];
  int k = threadIdx.x;
  float pkv = pk[k];
  float qk = fmaxf(pkv - diag[k] * invE, 0.0f);
  float t = qk + pkv;
  r0[k] = qk;
  r1[k] = qk * logf(qk + 1e-12f);
  r2[k] = t * logf(t + 1e-12f);
  __syncthreads();
  for (int s = 64; s > 0; s >>= 1) {
    if (k < s) { r0[k] += r0[k + s]; r1[k] += r1[k + s]; r2[k] += r2[k + s]; }
    __syncthreads();
  }
  if (k == 0) {
    float sq = r0[0];
    out[0] = sq * logf(sq + 1e-12f) - 2.0f * r1[0] - scal[0] + r2[0];
  }
}

// ---------------------------------------------------------------- launcher

extern "C" void kernel_launch(void* const* d_in, const int* in_sizes, int n_in,
                              void* d_out, int out_size, void* d_ws, size_t ws_size,
                              hipStream_t stream) {
  const float* node_features = (const float*)d_in[0];
  const float* rfp           = (const float*)d_in[1];
  const float* embed_w       = (const float*)d_in[2];
  const float* gamma         = (const float*)d_in[3];
  const float* beta          = (const float*)d_in[4];
  const float* rfp_w[3]  = {(const float*)d_in[5], (const float*)d_in[6], (const float*)d_in[7]};
  const float* rfp2_w[3] = {(const float*)d_in[8], (const float*)d_in[9], (const float*)d_in[10]};
  const float* dec_w[3]  = {(const float*)d_in[11], (const float*)d_in[12], (const float*)d_in[13]};
  const int* edges = (const int*)d_in[14];

  const int D = 128, K = 128;
  const int N = in_sizes[0] / D;
  const int B = in_sizes[1] / in_sizes[0];
  const int E = in_sizes[14] / 2;
  const int* srcI = edges;
  const int* dstI = edges + E;
  const int SPEC[3] = {128, 128, 64};
  const float invE = 1.0f / (float)E;

  // ---- workspace carving (deterministic; every region 32B aligned) ----
  float* ws = (float*)d_ws;
  size_t off = 0;
  auto alloc = [&](size_t n) { float* q = ws + off; off += n; return q; };
  float* deg_in = alloc(N);
  float* deg    = alloc(N);
  float* snorm  = alloc(N);
  float* pvec   = alloc(N);
  float* enorm  = alloc(E);
  float* diag   = alloc(K);     // diag, pk, scal contiguous -> one zero pass
  float* pk     = alloc(K);
  float* scal   = alloc(8);     // padded so bf16 pool stays 32B aligned
  __bf16* wbpool = (__bf16*)alloc(80000);  // 160000 bf16 slots (need 155648)
  size_t wboff = 0;
  auto wballoc = [&](size_t n) { __bf16* q = wbpool + wboff; wboff += n; return q; };
  __bf16* wb_embed  = wballoc((size_t)2 * D * 128);
  __bf16* wb_rfp[3]  = {wballoc(128 * 128), wballoc(128 * 128), wballoc(128 * 64)};
  __bf16* wb_rfp2[3] = {wballoc(128 * 128), wballoc(128 * 128), wballoc(128 * 64)};
  __bf16* wb_dec[3]  = {wballoc(64 * 128), wballoc(128 * 128), wballoc(128 * 128)};
  float* hbuf  = alloc((size_t)N * 128);
  float* traj  = alloc((size_t)B * N * 128);
  float* tsum  = alloc((size_t)N * 128);
  float* enc   = alloc((size_t)N * 64);
  float* nodeA = alloc((size_t)N * 128);
  (void)ws_size; (void)n_in; (void)out_size;

  float* Sout = (float*)d_out + (size_t)K * 64;                   // [N,128]
  float* lossp = (float*)d_out + (size_t)K * 64 + (size_t)N * K;  // scalar

  // ---- weight pack fp32 -> bf16 WMMA B layout ----
  auto pack = [&](const float* s, __bf16* d, int Kd, int F) {
    k_pack_w<<<CDIV(Kd * F, 256), 256, 0, stream>>>(s, d, Kd, F);
  };
  pack(embed_w, wb_embed, 2 * D, 128);
  int rin[3] = {128, 128, 128};
  for (int i = 0; i < 3; ++i) {
    pack(rfp_w[i],  wb_rfp[i],  rin[i], SPEC[i]);
    pack(rfp2_w[i], wb_rfp2[i], rin[i], SPEC[i]);
  }
  pack(dec_w[0], wb_dec[0], 64, 128);
  pack(dec_w[1], wb_dec[1], 128, 128);
  pack(dec_w[2], wb_dec[2], 128, 128);

  // ---- degrees & GCN norms ----
  k_zero<<<CDIV((size_t)N, 256), 256, 0, stream>>>(deg_in, (size_t)N);
  k_zero<<<2, 256, 0, stream>>>(diag, (size_t)(2 * K + 8));  // diag+pk+scal
  k_deg<<<CDIV(E, 256), 256, 0, stream>>>(dstI, deg_in, E);
  k_norm<<<CDIV(N, 256), 256, 0, stream>>>(deg_in, deg, snorm, pvec, N, invE);
  k_enorm<<<CDIV(E, 256), 256, 0, stream>>>(srcI, dstI, deg, enorm, E);

  // ---- embed + BN -> traj [B,N,128], traj_sum ----
  k_embed_gemm<<<dim3(B * N / 16), dim3(32, 8), 0, stream>>>(rfp, node_features, wb_embed,
                                                             gamma, beta, traj, N);
  k_sum_traj<<<CDIV((size_t)N * 128, 256), 256, 0, stream>>>(traj, tsum, (size_t)N * 128, B);

  // GCN conv: y = scatter(enorm * (x@W)[src] -> dst) + snorm * (x@W). In-place safe
  // (x fully consumed into hbuf by the GEMM before y is written).
  auto conv = [&](const float* x, const __bf16* w, float* y, int Kd, int F) {
    k_gemm<<<dim3(N / 16), dim3(32, F / 16), 0, stream>>>(x, w, hbuf, Kd, F);
    size_t tot4 = (size_t)N * F / 4;
    k_self_init4<<<CDIV(tot4, 256), 256, 0, stream>>>(hbuf, snorm, y, tot4, F / 4);
    k_scatter4<<<4096, 256, 0, stream>>>(srcI, dstI, enorm, hbuf, y, E, F / 4);
  };

  // ---- RFP layers ----
  const size_t bstride = (size_t)N * 128;
  int Kd = 128;
  for (int i = 0; i < 3; ++i) {
    int F = SPEC[i];
    for (int b = 0; b < B; ++b)
      conv(traj + (size_t)b * bstride, wb_rfp[i], traj + (size_t)b * bstride, Kd, F);
    conv(tsum, wb_rfp2[i], tsum, Kd, F);
    size_t NF = (size_t)N * F;
    if (i < 2) k_relu<<<CDIV(NF, 256), 256, 0, stream>>>(tsum, NF);
    k_add_bcast<<<CDIV(NF, 256), 256, 0, stream>>>(traj, tsum, NF, bstride, B, i < 2 ? 1 : 0);
    Kd = F;
  }

  // ---- node = (traj + traj_sum).sum(0) -> enc [N,64] ----
  k_finalize<<<CDIV((size_t)N * 64, 256), 256, 0, stream>>>(traj, tsum, enc, (size_t)N * 64,
                                                            bstride, B);

  // ---- decoder ----
  conv(enc, wb_dec[0], nodeA, 64, 128);
  k_relu<<<CDIV((size_t)N * 128, 256), 256, 0, stream>>>(nodeA, (size_t)N * 128);
  conv(nodeA, wb_dec[1], nodeA, 128, 128);
  k_relu<<<CDIV((size_t)N * 128, 256), 256, 0, stream>>>(nodeA, (size_t)N * 128);
  conv(nodeA, wb_dec[2], Sout, 128, 128);
  k_softmax128<<<N, 128, 0, stream>>>(Sout);

  // ---- pooled features out = S^T @ enc ----
  k_pool_gemm<<<dim3(8, 4), 32, 0, stream>>>(Sout, enc, (float*)d_out, N);

  // ---- map-equation loss ----
  k_diag<<<512, 128, 0, stream>>>(srcI, dstI, Sout, diag, E);
  k_pk<<<512, 128, 0, stream>>>(Sout, pvec, pk, N);
  k_plogp_sum<<<256, 256, 0, stream>>>(pvec, scal, N);
  k_loss<<<1, 128, 0, stream>>>(pk, diag, scal, lossp, invE);
}
